// HedgeHogAttention_81716047774352
// MI455X (gfx1250) — compile-verified, matrix operand order checked by reading, not run on hardware
//
#include <hip/hip_runtime.h>

typedef __attribute__((ext_vector_type(16))) _Float16 v16h;
typedef __attribute__((ext_vector_type(8)))  _Float16 v8h;
typedef __attribute__((ext_vector_type(8)))  float    v8f;
typedef __attribute__((ext_vector_type(4)))  float    v4f;

#define N_  2048
#define D_  64
#define D2_ 128
#define BH_ 32      // b*h = 2*16

// ---- WMMA fragment helpers (layouts per CDNA5 ISA 7.12.2) -------------------
// A (16x32 f16): lane L holds row M=L&15; half=L>>4; a[0..7]=K[8h..8h+7],
// a[8..15]=K[16+8h..16+8h+7].  B (32x16) is the dual: lane = column N,
// per-lane values gathered along K with the same index pattern, i.e. reading
// rows of the *transposed-in-memory* matrix contiguously along K.

__device__ __forceinline__ v16h frag_f16(const _Float16* __restrict__ base,
                                         int ld, int r0, int koff, int lane) {
  int r    = r0 + (lane & 15);
  int half = (lane >> 4) & 1;
  const _Float16* p = base + (size_t)r * ld + koff + 8 * half;
  v8h lo = *(const v8h*)(p);
  v8h hi = *(const v8h*)(p + 16);
  return __builtin_shufflevector(lo, hi, 0,1,2,3,4,5,6,7,8,9,10,11,12,13,14,15);
}

__device__ __forceinline__ v16h frag_f32(const float* __restrict__ base,
                                         int ld, int r0, int koff, int lane) {
  int r    = r0 + (lane & 15);
  int half = (lane >> 4) & 1;
  const float* p = base + (size_t)r * ld + koff + 8 * half;
  v16h o;
#pragma unroll
  for (int i = 0; i < 8; ++i) {
    o[i]     = (_Float16)p[i];
    o[8 + i] = (_Float16)p[16 + i];
  }
  return o;
}

// ---- Phase 1: projections + hedgehog features -------------------------------
// One wave handles one (bh, 16-row tile): q0/k0 = x@W^T (f16 out),
// fq/fk = [exp(h), exp(-h)], h = q0@Wm^T.
__global__ __launch_bounds__(128)
void hh_phase1(const float* __restrict__ x,
               const float* __restrict__ Wq,  const float* __restrict__ Wk,
               const float* __restrict__ Wmq, const float* __restrict__ Wmk,
               _Float16* __restrict__ q0h, _Float16* __restrict__ k0h,
               _Float16* __restrict__ fq,  _Float16* __restrict__ fk)
{
  __shared__ float stage[4][16][68];   // padded to dodge bank conflicts
  const int wave = threadIdx.x >> 5;
  const int lane = threadIdx.x & 31;
  const int t    = blockIdx.x * 4 + wave;   // 0..4095
  const int bh   = t >> 7;
  const int m0   = (t & 127) << 4;
  const int half = (lane >> 4) & 1;
  const int nco  = lane & 15;

  const float* xp = x + ((size_t)bh * N_ + m0) * D_;
  v16h ax0 = frag_f32(xp, D_, 0, 0,  lane);
  v16h ax1 = frag_f32(xp, D_, 0, 32, lane);

  float (*st)[68] = stage[wave];

#pragma unroll
  for (int path = 0; path < 2; ++path) {
    const float* W    = path ? Wk  : Wq;
    const float* Wm   = path ? Wmk : Wmq;
    _Float16*    proj = path ? k0h : q0h;
    _Float16*    feat = path ? fk  : fq;

    // projection: p0[16x64] = x_tile @ W^T
#pragma unroll
    for (int e = 0; e < 4; ++e) {
      v16h b0 = frag_f32(W, D_, e * 16, 0,  lane);
      v16h b1 = frag_f32(W, D_, e * 16, 32, lane);
      v8f acc = {};
      acc = __builtin_amdgcn_wmma_f32_16x16x32_f16(false, ax0, false, b0, (short)0, acc, false, false);
      acc = __builtin_amdgcn_wmma_f32_16x16x32_f16(false, ax1, false, b1, (short)0, acc, false, false);
#pragma unroll
      for (int r = 0; r < 8; ++r)
        st[8 * half + r][e * 16 + nco] = acc[r];
    }
    asm volatile("s_wait_dscnt 0x0" ::: "memory");  // in-wave LDS RAW

    // spill projection as f16 for phase 2
#pragma unroll
    for (int idx = lane; idx < 16 * 64; idx += 32) {
      int row = idx >> 6, col = idx & 63;
      proj[((size_t)bh * N_ + m0 + row) * D_ + col] = (_Float16)st[row][col];
    }

    // re-fragment projection (f32 LDS tile -> f16 A-operand)
    v16h ap0, ap1;
    {
      int r = lane & 15;
#pragma unroll
      for (int i = 0; i < 8; ++i) {
        ap0[i]     = (_Float16)st[r][8 * half + i];
        ap0[8 + i] = (_Float16)st[r][16 + 8 * half + i];
        ap1[i]     = (_Float16)st[r][32 + 8 * half + i];
        ap1[8 + i] = (_Float16)st[r][48 + 8 * half + i];
      }
    }

    // feature GEMM h = p0 @ Wm^T, then feat = [exp(h), exp(-h)]
#pragma unroll
    for (int e = 0; e < 4; ++e) {
      v16h b0 = frag_f32(Wm, D_, e * 16, 0,  lane);
      v16h b1 = frag_f32(Wm, D_, e * 16, 32, lane);
      v8f acc = {};
      acc = __builtin_amdgcn_wmma_f32_16x16x32_f16(false, ap0, false, b0, (short)0, acc, false, false);
      acc = __builtin_amdgcn_wmma_f32_16x16x32_f16(false, ap1, false, b1, (short)0, acc, false, false);
#pragma unroll
      for (int r = 0; r < 8; ++r) {
        float hv  = acc[r];
        int   row = m0 + 8 * half + r;
        int   col = e * 16 + nco;
        _Float16* fp = feat + ((size_t)bh * N_ + row) * D2_;
        fp[col]      = (_Float16)__expf(hv);
        fp[col + D_] = (_Float16)__expf(-hv);
      }
    }
  }
}

// ---- Phase 2: attention maps, normalization, single-pass output write -------
// One workgroup (8 waves) = one (bh, 16-row tile) against all 2048 columns.
// Both 16x2048 f32 score tiles are staged in LDS (256 KB < 320 KB/WGP).
__global__ __launch_bounds__(256)
void hh_phase2(const _Float16* __restrict__ q0h, const _Float16* __restrict__ k0h,
               const _Float16* __restrict__ fq,  const _Float16* __restrict__ fk,
               float* __restrict__ out)
{
  __shared__ float s1[16][N_];
  __shared__ float s2[16][N_];
  __shared__ float part1[16][16];
  __shared__ float part2[16][16];
  __shared__ float rinv1[16], rinv2[16];

  const int tid  = threadIdx.x;
  const int wave = tid >> 5;
  const int lane = tid & 31;
  const int bh   = blockIdx.x >> 7;
  const int m0   = (blockIdx.x & 127) << 4;
  const int half = (lane >> 4) & 1;
  const int nco  = lane & 15;

  // A-operands for this 16-row tile (kept in registers across all n-tiles)
  const _Float16* fqp = fq  + ((size_t)bh * N_ + m0) * D2_;
  const _Float16* qp  = q0h + ((size_t)bh * N_ + m0) * D_;
  v16h afq[4], aq[2];
#pragma unroll
  for (int kc = 0; kc < 4; ++kc) afq[kc] = frag_f16(fqp, D2_, 0, kc * 32, lane);
#pragma unroll
  for (int kc = 0; kc < 2; ++kc) aq[kc]  = frag_f16(qp,  D_,  0, kc * 32, lane);

  const _Float16* fkb = fk  + (size_t)bh * N_ * D2_;
  const _Float16* kb  = k0h + (size_t)bh * N_ * D_;

  for (int nt = wave; nt < N_ / 16; nt += 8) {
    int n0 = nt << 4;
    v8f a1 = {};
#pragma unroll
    for (int kc = 0; kc < 4; ++kc) {
      v16h bfrag = frag_f16(fkb, D2_, n0, kc * 32, lane);  // B[k,n] = fk[n,k]
      a1 = __builtin_amdgcn_wmma_f32_16x16x32_f16(false, afq[kc], false, bfrag, (short)0, a1, false, false);
    }
    v8f a2 = {};
#pragma unroll
    for (int kc = 0; kc < 2; ++kc) {
      v16h bfrag = frag_f16(kb, D_, n0, kc * 32, lane);    // B[k,n] = k0[n,k]
      a2 = __builtin_amdgcn_wmma_f32_16x16x32_f16(false, aq[kc], false, bfrag, (short)0, a2, false, false);
    }
    int col = n0 + nco;
#pragma unroll
    for (int r = 0; r < 8; ++r) {
      int row = 8 * half + r;
      s1[row][col] = a1[r];                         // fq.fk^T  (positive)
      s2[row][col] = __expf(a2[r] * 0.125f);        // exp(q.k / sqrt(64))
    }
  }
  __syncthreads();

  // row sums over all 2048 columns (16 threads per row)
  {
    int row = tid & 15, seg = tid >> 4;
    float sa = 0.f, sb = 0.f;
    int c0 = seg * (N_ / 16);
    for (int c = 0; c < N_ / 16; ++c) { sa += s1[row][c0 + c]; sb += s2[row][c0 + c]; }
    part1[row][seg] = sa; part2[row][seg] = sb;
  }
  __syncthreads();
  if (tid < 16) {
    float sa = 0.f, sb = 0.f;
#pragma unroll
    for (int s = 0; s < 16; ++s) { sa += part1[tid][s]; sb += part2[tid][s]; }
    rinv1[tid] = 1.0f / sa;
    rinv2[tid] = 1.0f / sb;
  }
  __syncthreads();

  // normalized, coalesced float4 writes; output touched exactly once
  size_t predBase = ((size_t)bh * N_ + m0) * N_;
  size_t trueBase = predBase + (size_t)BH_ * N_ * N_;
  for (int idx = tid; idx < 16 * (N_ / 4); idx += 256) {
    int row = idx >> 9;          // N_/4 = 512 vec4 per row
    int c4  = (idx & 511) << 2;
    v4f v1 = *(const v4f*)&s1[row][c4];
    v4f v2 = *(const v4f*)&s2[row][c4];
    v1 *= rinv1[row];
    v2 *= rinv2[row];
    *(v4f*)(out + predBase + (size_t)row * N_ + c4) = v1;
    *(v4f*)(out + trueBase + (size_t)row * N_ + c4) = v2;
  }
}

// ---- host glue --------------------------------------------------------------
extern "C" void kernel_launch(void* const* d_in, const int* in_sizes, int n_in,
                              void* d_out, int out_size, void* d_ws, size_t ws_size,
                              hipStream_t stream) {
  (void)in_sizes; (void)n_in; (void)out_size; (void)ws_size;
  const float* x   = (const float*)d_in[0];
  const float* Wq  = (const float*)d_in[1];
  const float* Wk  = (const float*)d_in[3];
  const float* Wmq = (const float*)d_in[5];
  const float* Wmk = (const float*)d_in[7];
  float* out = (float*)d_out;

  // f16 workspace: q0 | k0 | fq | fk  (48 MB total)
  _Float16* ws  = (_Float16*)d_ws;
  _Float16* q0h = ws;
  _Float16* k0h = q0h + (size_t)BH_ * N_ * D_;
  _Float16* fqw = k0h + (size_t)BH_ * N_ * D_;
  _Float16* fkw = fqw + (size_t)BH_ * N_ * D2_;

  hh_phase1<<<dim3(1024), dim3(128), 0, stream>>>(x, Wq, Wk, Wmq, Wmk, q0h, k0h, fqw, fkw);
  hh_phase2<<<dim3(BH_ * (N_ / 16)), dim3(256), 0, stream>>>(q0h, k0h, fqw, fkw, out);
}
